// Caterpillar_90829968376042
// MI455X (gfx1250) — matrix operand order repaired and checked by reading, not compiled, not run on hardware
//
#include <hip/hip_runtime.h>
#include <hip/hip_bf16.h>

// Problem constants (match reference)
#define NB   2
#define TT   2048
#define DMD  1024
#define HH   8
#define EE   4
#define LL   32
#define DKV  64
#define PDIM 1568            // 512 Q | 512 K | 512 V | 32 G
#define MROWS (NB * TT)      // 4096
#define RECT  (TT + LL)      // 2080

typedef __attribute__((ext_vector_type(16))) __bf16 v16bf;
typedef __attribute__((ext_vector_type(8)))  float  v8f;

union FragBF { uint4 q[2]; v16bf v; };

__device__ __forceinline__ unsigned short f2bf(float x) {
  unsigned int u = __float_as_uint(x);
  u += 0x7FFFu + ((u >> 16) & 1u);           // round-to-nearest-even
  return (unsigned short)(u >> 16);
}

// ---------------------------------------------------------------------------
// Kernel 1: pack X -> bf16, weights -> concatenated bf16 B-matrices
// ---------------------------------------------------------------------------
__global__ void k_prep(const float* __restrict__ X,
                       const float* __restrict__ wG,
                       const float* __restrict__ wK,
                       const float* __restrict__ wV,
                       const float* __restrict__ wQ,
                       const float* __restrict__ wO,
                       unsigned short* __restrict__ Xbf,
                       unsigned short* __restrict__ Wcat,
                       unsigned short* __restrict__ WoT) {
  const long nX  = (long)MROWS * DMD;
  const long nW  = (long)PDIM * DMD;
  const long nWo = (long)DMD * 512;
  const long total = nX + nW + nWo;
  for (long i = blockIdx.x * (long)blockDim.x + threadIdx.x; i < total;
       i += (long)gridDim.x * blockDim.x) {
    if (i < nX) {
      Xbf[i] = f2bf(X[i]);
    } else if (i < nX + nW) {
      long j = i - nX;
      int p = (int)(j / DMD);
      int c = (int)(j % DMD);
      float v;
      if (p < 512)       v = wQ[(long)p * DMD + c];
      else if (p < 1024) v = wK[(long)(p - 512) * DMD + c];
      else if (p < 1536) v = wV[(long)(p - 1024) * DMD + c];
      else               v = wG[(long)(p - 1536) * DMD + c];
      Wcat[j] = f2bf(v);
    } else {
      long j = i - nX - nW;
      int nn = (int)(j / 512);
      int k  = (int)(j % 512);
      WoT[j] = f2bf(wO[(long)k * DMD + nn]);   // transpose w_O
    }
  }
}

// ---------------------------------------------------------------------------
// Kernel 2/6: bf16 WMMA GEMM.  C[M,NN] = A[M,K] * B[NN,K]^T
// Each wave owns a 64x32 C tile: 4 M-subtiles x 2 N-subtiles,
// 8x v_wmma_f32_16x16x32_bf16 per 32-wide k-step.
// Fragments are hand-packed per CDNA5 ISA 7.12.2 wave32 layouts; both A and B
// fragments are exactly two contiguous 16-byte loads per lane:
//   A frag: [m*K + k0 + hi*8 .. +7] and [m*K + k0 + 16 + hi*8 .. +7]
//   B frag: [n*K + k0 + hi*8 .. +7] and [n*K + k0 + 16 + hi*8 .. +7]
// ---------------------------------------------------------------------------
__global__ void k_gemm(const unsigned short* __restrict__ A,
                       const unsigned short* __restrict__ B,
                       float* __restrict__ C,
                       int M, int K, int NN, int totalWaves) {
  const int wave = blockIdx.x * (blockDim.x >> 5) + (threadIdx.x >> 5);
  if (wave >= totalWaves) return;
  const int lane = threadIdx.x & 31;
  const int ntiles = NN >> 5;
  const int mb = (wave / ntiles) * 64;
  const int nb = (wave % ntiles) * 32;
  const int l16 = lane & 15, hi = lane >> 4;

  v8f acc[4][2] = {{{}, {}}, {{}, {}}, {{}, {}}, {{}, {}}};

  const unsigned short* pa[4];
#pragma unroll
  for (int i = 0; i < 4; ++i)
    pa[i] = A + (long)(mb + 16 * i + l16) * K + hi * 8;
  const unsigned short* pb[2];
#pragma unroll
  for (int j = 0; j < 2; ++j)
    pb[j] = B + (long)(nb + 16 * j + l16) * K + hi * 8;

  for (int k0 = 0; k0 < K; k0 += 32) {
    FragBF a[4], b[2];
#pragma unroll
    for (int i = 0; i < 4; ++i) {
      a[i].q[0] = *(const uint4*)(pa[i] + k0);
      a[i].q[1] = *(const uint4*)(pa[i] + k0 + 16);
    }
#pragma unroll
    for (int j = 0; j < 2; ++j) {
      b[j].q[0] = *(const uint4*)(pb[j] + k0);
      b[j].q[1] = *(const uint4*)(pb[j] + k0 + 16);
    }
    // prefetch next k-tile (global_prefetch_b8); speculative, OOB is dropped
    __builtin_prefetch(pa[0] + k0 + 32, 0, 1);
    __builtin_prefetch(pb[0] + k0 + 32, 0, 1);
#pragma unroll
    for (int i = 0; i < 4; ++i)
#pragma unroll
      for (int j = 0; j < 2; ++j)
        acc[i][j] = __builtin_amdgcn_wmma_f32_16x16x32_bf16(
            false, a[i].v, false, b[j].v, (short)0, acc[i][j], false, false);
  }

  // f32 C/D 16x16 layout: col = lane%16, row = vgpr + 8*(lane/16)
  const int n0 = nb + l16, n1 = nb + 16 + l16;
#pragma unroll
  for (int i = 0; i < 4; ++i) {
#pragma unroll
    for (int v = 0; v < 8; ++v) {
      int m = mb + 16 * i + v + 8 * hi;
      C[(long)m * NN + n0] = acc[i][0][v];
      C[(long)m * NN + n1] = acc[i][1][v];
    }
  }
}

// ---------------------------------------------------------------------------
// Kernel 3: gates.  One block per (n,e,t), 128 threads over [K(64)|V(64)].
// gated[n,e,t,0:64] = sum_h sig(G)*K ; [64:128] = sum_h sig(G)*V ; A = 1-sum G
// ---------------------------------------------------------------------------
__global__ void k_gate(const float* __restrict__ QKVG,
                       const float* __restrict__ bG,
                       float* __restrict__ Abuf,
                       float* __restrict__ gated) {
  __shared__ float gs[HH];
  int b = blockIdx.x;
  int t = b % TT;
  int e = (b / TT) % EE;
  int n = b / (TT * EE);
  int nt = n * TT + t;
  const float* row = QKVG + (long)nt * PDIM;
  int tid = threadIdx.x;
  if (tid < HH) {
    float x = row[1536 + tid * EE + e] + bG[tid * EE + e];
    gs[tid] = 1.0f / (1.0f + __expf(-x));
  }
  __syncthreads();
  int off = (tid < 64) ? (512 + tid) : (1024 + (tid - 64));
  float acc = 0.0f;
#pragma unroll
  for (int h = 0; h < HH; ++h) acc += gs[h] * row[off + h * 64];
  long gbase = (long)(n * EE + e) * TT + t;
  gated[gbase * 128 + tid] = acc;
  if (tid == 0) {
    float s = 0.0f;
#pragma unroll
    for (int h = 0; h < HH; ++h) s += gs[h];
    Abuf[gbase] = 1.0f - s;
  }
}

// ---------------------------------------------------------------------------
// Kernel 4: cross-chunk recurrence.  One thread per (n,e,l,d128); 64 steps.
// rec[(n,e), s, d] : s<L -> init ; s=L+tau -> scan output at time tau.
// ---------------------------------------------------------------------------
__global__ void k_scan(const float* __restrict__ Abuf,
                       const float* __restrict__ gated,
                       const float* __restrict__ initK,
                       const float* __restrict__ initV,
                       float* __restrict__ rec) {
  int id = blockIdx.x * blockDim.x + threadIdx.x;   // 32768 total
  int d = id & 127;
  int l = (id >> 7) & 31;
  int e = (id >> 12) & 3;
  int n = id >> 14;
  float y = (d < 64) ? initK[(e * LL + l) * DKV + d]
                     : initV[(e * LL + l) * DKV + (d - 64)];
  long ne = (long)(n * EE + e);
  rec[(ne * RECT + l) * 128 + d] = y;
  long abase = ne * TT;
#pragma unroll 1
  for (int m = 0; m < TT / LL; ++m) {
    int tau = m * LL + l;
    float a = Abuf[abase + tau];
    float g = gated[(abase + tau) * 128 + d];
    y = a * y + g;
    rec[(ne * RECT + LL + tau) * 128 + d] = y;
  }
}

// ---------------------------------------------------------------------------
// Kernel 5: sliding attention.  One block per (n,t), 128 threads.
// Slots: s = e*32 + l, key/value = rec[(n,e), t+l+1, 0:64 / 64:128].
// ---------------------------------------------------------------------------
__global__ void k_attn(const float* __restrict__ QKVG,
                       const float* __restrict__ rec,
                       unsigned short* __restrict__ Ybf) {
  __shared__ float sq[HH * DKV];      // 512  queries, all heads
  __shared__ float sv[128 * DKV];     // 8192 values (reused 8x across heads)
  __shared__ float sc[HH * 128];      // scores -> probabilities
  __shared__ float smx[HH];
  __shared__ float sdn[HH];
  int b = blockIdx.x;
  int t = b % TT;
  int n = b / TT;
  int nt = n * TT + t;
  int tid = threadIdx.x;
  const float* qrow = QKVG + (long)nt * PDIM;     // Q at offset 0: h*64+d
  for (int i = tid; i < HH * DKV; i += 128) sq[i] = qrow[i];
  for (int i = tid; i < 128 * DKV; i += 128) {
    int s = i >> 6, d = i & 63;
    int e = s >> 5, l = s & 31;
    sv[i] = rec[((long)(n * EE + e) * RECT + (t + l + 1)) * 128 + 64 + d];
  }
  __syncthreads();
  {
    int s = tid;
    int e = s >> 5, l = s & 31;
    const float* krow = rec + ((long)(n * EE + e) * RECT + (t + l + 1)) * 128;
    float kreg[DKV];
#pragma unroll
    for (int d = 0; d < DKV; ++d) kreg[d] = krow[d];
#pragma unroll
    for (int h = 0; h < HH; ++h) {
      float acc = 0.0f;
#pragma unroll
      for (int d = 0; d < DKV; ++d) acc += sq[h * DKV + d] * kreg[d];
      sc[h * 128 + s] = acc * 0.125f;             // 1/sqrt(64)
    }
  }
  __syncthreads();
  if (tid < HH) {
    float mx = -3.0e38f;
    for (int s = 0; s < 128; ++s) mx = fmaxf(mx, sc[tid * 128 + s]);
    smx[tid] = mx;
  }
  __syncthreads();
#pragma unroll
  for (int h = 0; h < HH; ++h)
    sc[h * 128 + tid] = __expf(sc[h * 128 + tid] - smx[h]);
  __syncthreads();
  if (tid < HH) {
    float sum = 0.0f;
    for (int s = 0; s < 128; ++s) sum += sc[tid * 128 + s];
    sdn[tid] = 1.0f / sum;
  }
  __syncthreads();
#pragma unroll
  for (int i = 0; i < 4; ++i) {
    int o = tid + i * 128;        // o = h*64 + d
    int h = o >> 6, d = o & 63;
    float acc = 0.0f;
    for (int s = 0; s < 128; ++s) acc += sc[h * 128 + s] * sv[s * DKV + d];
    Ybf[(long)nt * 512 + o] = f2bf(acc * sdn[h]);
  }
}

// ---------------------------------------------------------------------------
extern "C" void kernel_launch(void* const* d_in, const int* in_sizes, int n_in,
                              void* d_out, int out_size, void* d_ws, size_t ws_size,
                              hipStream_t stream) {
  (void)in_sizes; (void)n_in; (void)out_size; (void)ws_size;
  const float* X  = (const float*)d_in[0];
  const float* wG = (const float*)d_in[1];
  const float* bG = (const float*)d_in[2];
  const float* wK = (const float*)d_in[3];
  const float* wV = (const float*)d_in[4];
  const float* wQ = (const float*)d_in[5];
  const float* wO = (const float*)d_in[6];
  const float* iK = (const float*)d_in[7];
  const float* iV = (const float*)d_in[8];
  float* out = (float*)d_out;

  char* p = (char*)d_ws;
  auto take = [&](size_t bytes) {
    char* r = p;
    p += (bytes + 255) & ~(size_t)255;
    return r;
  };
  unsigned short* Xbf  = (unsigned short*)take((size_t)MROWS * DMD * 2);
  unsigned short* Wcat = (unsigned short*)take((size_t)PDIM * DMD * 2);
  unsigned short* WoT  = (unsigned short*)take((size_t)DMD * 512 * 2);
  unsigned short* Ybf  = (unsigned short*)take((size_t)MROWS * 512 * 2);
  float* QKVG  = (float*)take((size_t)MROWS * PDIM * 4);
  float* Abuf  = (float*)take((size_t)NB * EE * TT * 4);
  float* gated = (float*)take((size_t)NB * EE * TT * 128 * 4);
  float* rec   = (float*)take((size_t)NB * EE * RECT * 128 * 4);

  k_prep<<<4096, 256, 0, stream>>>(X, wG, wK, wV, wQ, wO, Xbf, Wcat, WoT);

  {
    int waves = (MROWS / 64) * (PDIM / 32);      // 64 * 49 = 3136
    k_gemm<<<waves / 4, 128, 0, stream>>>(Xbf, Wcat, QKVG, MROWS, DMD, PDIM, waves);
  }

  k_gate<<<NB * EE * TT, 128, 0, stream>>>(QKVG, bG, Abuf, gated);
  k_scan<<<128, 256, 0, stream>>>(Abuf, gated, iK, iV, rec);
  k_attn<<<NB * TT, 128, 0, stream>>>(QKVG, rec, Ybf);

  {
    int waves = (MROWS / 64) * (DMD / 32);       // 64 * 32 = 2048
    k_gemm<<<waves / 4, 128, 0, stream>>>(Ybf, WoT, out, MROWS, 512, DMD, waves);
  }
}
